// SeqCondAttention_45346264711371
// MI455X (gfx1250) — matrix operand order repaired and checked by reading, not compiled, not run on hardware
//
#include <hip/hip_runtime.h>
#include <hip/hip_bf16.h>

// ---------------------------------------------------------------------------
// SeqCondAttention on MI455X (gfx1250).
// All GEMM operands materialized as bf16 (halves operand traffic, removes
// cvt from hot loop); weights pre-transposed+padded so the GEMM hot loop is
// pure 16B async global->LDS copies + ds_load_b128 fragments + v_wmma.
// ---------------------------------------------------------------------------

typedef __attribute__((ext_vector_type(16))) __bf16 v16bf;
typedef __attribute__((ext_vector_type(8)))  float  v8f;
typedef __attribute__((ext_vector_type(4)))  int    v4i;

#define BB        4
#define LL        4096
#define DD        768
#define KK        12
#define NREP      2
#define HH        64
#define CONVK     4
#define DIM_MEM   768
#define DIM_MEMK  780
#define DIM_QTOT  768
#define DIM_LAT   192
#define DIM_AUX   972
#define DIM_SWH   384
#define C_STACK   1548
#define DIM_YPRE  2304
#define NTOK      (BB * LL)      // 16384

#define NPAD_MEM  896            // 780 -> 7*128
#define NPAD_AUX  1024           // 972 -> 8*128

#define GLOBAL_AS __attribute__((address_space(1)))
#define LDS_AS    __attribute__((address_space(3)))

// 16B global -> LDS copy: async DMA when available (ASYNCcnt), else direct.
__device__ __forceinline__ void cp16_g2l(const void* g, void* l) {
#if defined(__HIP_DEVICE_COMPILE__) && __has_builtin(__builtin_amdgcn_global_load_async_to_lds_b128)
    __builtin_amdgcn_global_load_async_to_lds_b128(
        (GLOBAL_AS v4i*)const_cast<void*>(g),
        (LDS_AS v4i*)l, 0, 0);
#else
    *(uint4*)l = *(const uint4*)g;
#endif
}

#if defined(__HIP_DEVICE_COMPILE__)
#  if __has_builtin(__builtin_amdgcn_s_wait_asynccnt)
#    define WAIT_ASYNC(n) __builtin_amdgcn_s_wait_asynccnt(n)
#  else
#    define WAIT_ASYNC(n) asm volatile("s_wait_asynccnt %0" :: "i"(n) : "memory")
#  endif
#else
#  define WAIT_ASYNC(n) ((void)0)
#endif

__device__ __forceinline__ float softplus_f(float x) {
    return (x > 20.f) ? x : log1pf(__expf(x));
}
__device__ __forceinline__ float sigmoid_f(float x) {
    return 1.f / (1.f + __expf(-x));
}

// ---------------------------------------------------------------------------
// GEMM: C(f32) = A(bf16, MxK row-major) * Bt(bf16, NpadxK row-major, i.e. B
// transposed and zero-padded in N).  Requirements: M % 64 == 0, K % 32 == 0,
// grid.x == Npad/128 (B rows always valid); only C stores guard real Ndim.
// Block = 256 threads = 8 waves.  BM=64, BN=128, BK=32, double-buffered LDS,
// async copies (3 per wave per stage), 4 v_wmma per wave per K-step.
// ---------------------------------------------------------------------------
__global__ __launch_bounds__(256)
void gemm_bf16_wmma(const __bf16* __restrict__ A, long lda, long strideA,
                    const __bf16* __restrict__ Bt, long ldbt, long strideB,
                    float* __restrict__ C, long ldc, long strideC,
                    int Ndim, int Kdim)
{
    constexpr int BM = 64, BN = 128, BK = 32, PITCH = 40; // 80B pitch (16B mult)
    __shared__ __bf16 As[2][BM][PITCH];   // [m][k]
    __shared__ __bf16 Bs[2][BN][PITCH];   // [n][k]

    const __bf16* Ab = A  + (long)blockIdx.z * strideA;
    const __bf16* Bb = Bt + (long)blockIdx.z * strideB;
    float*        Cb = C  + (long)blockIdx.z * strideC;

    const int tileM = blockIdx.y * BM;
    const int tileN = blockIdx.x * BN;
    const int tid   = threadIdx.x;
    const int wave  = tid >> 5;
    const int lane  = tid & 31;
    const int wr    = wave >> 1;   // 16-row strip
    const int wc    = wave & 1;    // 64-col strip
    const int half  = lane >> 4;
    const int l16   = lane & 15;

    // copy mapping: 16B chunks (8 bf16)
    const int crow = tid >> 2;            // 0..63
    const int cseg = (tid & 3) * 8;       // element offset of 16B segment
    const __bf16* a_g  = Ab + (long)(tileM + crow) * lda  + cseg;
    const __bf16* b_g0 = Bb + (long)(tileN + crow) * ldbt + cseg;
    const __bf16* b_g1 = Bb + (long)(tileN + 64 + crow) * ldbt + cseg;

    v8f acc[4] = {v8f{}, v8f{}, v8f{}, v8f{}};
    const int S = Kdim / BK;

    // stage 0
    cp16_g2l(a_g,  &As[0][crow][cseg]);
    cp16_g2l(b_g0, &Bs[0][crow][cseg]);
    cp16_g2l(b_g1, &Bs[0][64 + crow][cseg]);

    for (int s = 0; s < S; ++s) {
        const int cur = s & 1;
        if (s + 1 < S) {                    // issue next stage, keep it in flight
            const long k0 = (long)(s + 1) * BK;
            cp16_g2l(a_g  + k0, &As[cur ^ 1][crow][cseg]);
            cp16_g2l(b_g0 + k0, &Bs[cur ^ 1][crow][cseg]);
            cp16_g2l(b_g1 + k0, &Bs[cur ^ 1][64 + crow][cseg]);
            WAIT_ASYNC(3);
        } else {
            WAIT_ASYNC(0);
        }
        __syncthreads();

        union Frag { v16bf v; uint4 u[2]; };
        Frag af;
        const int m_l = wr * 16 + l16;
        af.u[0] = *(const uint4*)&As[cur][m_l][8 * half];
        af.u[1] = *(const uint4*)&As[cur][m_l][16 + 8 * half];

        #pragma unroll
        for (int t = 0; t < 4; ++t) {
            Frag bf;
            const int n0 = wc * 64 + t * 16 + l16;
            bf.u[0] = *(const uint4*)&Bs[cur][n0][8 * half];
            bf.u[1] = *(const uint4*)&Bs[cur][n0][16 + 8 * half];
            acc[t] = __builtin_amdgcn_wmma_f32_16x16x32_bf16(
                false, af.v, false, bf.v, (short)0, acc[t], false, false);
        }
        __syncthreads();
    }

    // store C (VGPR v: M = v + 8*half, N = lane%16)
    #pragma unroll
    for (int t = 0; t < 4; ++t) {
        const int gn = tileN + wc * 64 + t * 16 + l16;
        if (gn < Ndim) {
            const int gm0 = tileM + wr * 16 + (half << 3);
            #pragma unroll
            for (int v = 0; v < 8; ++v)
                Cb[(long)(gm0 + v) * ldc + gn] = acc[t][v];
        }
    }
}

// ---------------------------------------------------------------------------
// fp32 -> bf16 elementwise convert.
// ---------------------------------------------------------------------------
__global__ void cvt_bf16_kernel(const float* __restrict__ src,
                                __bf16* __restrict__ dst, long n)
{
    long idx = (long)blockIdx.x * blockDim.x + threadIdx.x;
    if (idx < n) dst[idx] = (__bf16)src[idx];
}

// ---------------------------------------------------------------------------
// Weight convert + transpose + pad: W(batch, Krows x Ncols) fp32 ->
// Wt(batch, Npad x Krows) bf16, rows n >= Ncols zero-filled.
// ---------------------------------------------------------------------------
__global__ void wt_transpose_kernel(const float* __restrict__ W,
                                    __bf16* __restrict__ Wt,
                                    int Krows, int Ncols, int Npad)
{
    long per = (long)Npad * Krows;
    long idx = (long)blockIdx.x * blockDim.x + threadIdx.x;
    if (idx >= per) return;
    const float* Wb  = W  + (long)blockIdx.z * ((long)Krows * Ncols);
    __bf16*      Wtb = Wt + (long)blockIdx.z * per;
    int  kr = (int)(idx % Krows);
    long n  = idx / Krows;
    float v = (n < Ncols) ? Wb[(long)kr * Ncols + n] : 0.f;
    Wtb[idx] = (__bf16)v;
}

// ---------------------------------------------------------------------------
// c_skip slice of z_aux -> contiguous bf16 (16384 x 192).
// ---------------------------------------------------------------------------
__global__ void cvt_cskip_kernel(const float* __restrict__ z_aux,
                                 __bf16* __restrict__ cskip)
{
    long idx = (long)blockIdx.x * blockDim.x + threadIdx.x;
    long total = (long)NTOK * DIM_LAT;
    if (idx >= total) return;
    long token = idx / DIM_LAT;
    int  j     = (int)(idx % DIM_LAT);
    cskip[idx] = (__bf16)z_aux[token * DIM_AUX + DIM_QTOT + j];
}

// ---------------------------------------------------------------------------
// Depthwise causal conv over L (left pad CONV_K-1).
// ---------------------------------------------------------------------------
__global__ void conv_dw_kernel(const float* __restrict__ z,
                               const float* __restrict__ ker,
                               float* __restrict__ out)
{
    long idx = (long)blockIdx.x * blockDim.x + threadIdx.x;
    long total = (long)NTOK * DIM_MEMK;
    if (idx >= total) return;
    int  c  = (int)(idx % DIM_MEMK);
    long bl = idx / DIM_MEMK;
    int  l  = (int)(bl % LL);
    long b  = bl / LL;
    float acc = 0.f;
    #pragma unroll
    for (int t = 0; t < CONVK; ++t) {
        int ls = l - (CONVK - 1) + t;
        if (ls >= 0)
            acc += ker[t * DIM_MEMK + c] * z[(b * LL + ls) * (long)DIM_MEMK + c];
    }
    out[idx] = acc;
}

// ---------------------------------------------------------------------------
// RMSNorm of q slice of z_aux; block (256 thr) per token.
// ---------------------------------------------------------------------------
__global__ __launch_bounds__(256)
void rms_q_kernel(const float* __restrict__ z_aux,
                  const float* __restrict__ rms_scale,
                  float* __restrict__ q)
{
    const long token = blockIdx.x;
    const float* row = z_aux + token * DIM_AUX;
    __shared__ float red[8];
    __shared__ float s_inv;

    float ss = 0.f;
    for (int i = threadIdx.x; i < DIM_QTOT; i += 256) {
        float v = row[i];
        ss += v * v;
    }
    #pragma unroll
    for (int off = 16; off > 0; off >>= 1)
        ss += __shfl_down(ss, off, 32);
    if ((threadIdx.x & 31) == 0) red[threadIdx.x >> 5] = ss;
    __syncthreads();
    if (threadIdx.x == 0) {
        float t = 0.f;
        #pragma unroll
        for (int i = 0; i < 8; ++i) t += red[i];
        s_inv = rsqrtf(t * (1.f / (float)DIM_QTOT) + 1e-6f);
    }
    __syncthreads();
    const float inv = s_inv;
    for (int i = threadIdx.x; i < DIM_QTOT; i += 256)
        q[token * DIM_QTOT + i] = row[i] * inv * rms_scale[i];
}

// ---------------------------------------------------------------------------
// Build "stack" = [p_w(K), re(K*H), im(K*H)] per token.  Thread per (b,l,k,h).
// ---------------------------------------------------------------------------
__global__ void stack_build_kernel(const float* __restrict__ zconv,
                                   const float* __restrict__ theta_raw,
                                   const float* __restrict__ tanh_scale,
                                   const float* __restrict__ decay_slopes,
                                   const float* __restrict__ score_scale,
                                   float* __restrict__ stack)
{
    long idx = (long)blockIdx.x * blockDim.x + threadIdx.x;
    long total = (long)NTOK * KK * HH;
    if (idx >= total) return;
    int  h     = (int)(idx % HH);
    int  k     = (int)((idx / HH) % KK);
    long token = idx / (KK * HH);
    int  l     = (int)(token % LL);

    const float* zc = zconv + token * DIM_MEMK;
    float s_raw = zc[DIM_MEM + k];
    float slope = softplus_f(decay_slopes[k]);
    float ltw   = -slope * (float)(LL - 1 - l);
    float lp    = fminf(fmaxf(score_scale[k] * s_raw, -20.f), 20.f);
    float p_w   = __expf(lp + ltw);

    float kv    = zc[k * HH + h];
    float theta = softplus_f(theta_raw[k * HH + h]) + 1e-3f;
    float phi   = tanhf(kv * tanh_scale[k]) * theta;
    float kvw   = kv * p_w;
    float s, c;
    __sincosf(phi, &s, &c);

    float* st = stack + token * C_STACK;
    st[KK + k * HH + h]           = kvw * c;   // re
    st[KK + DIM_MEM + k * HH + h] = kvw * s;   // im
    if (h == 0) st[k] = p_w;
}

// ---------------------------------------------------------------------------
// In-place inclusive cumsum along L; one thread per (batch, channel).
// ---------------------------------------------------------------------------
__global__ void scan_kernel(float* __restrict__ stack)
{
    long tid = (long)blockIdx.x * blockDim.x + threadIdx.x;
    if (tid >= (long)BB * C_STACK) return;
    long b = tid / C_STACK;
    long c = tid % C_STACK;
    float acc = 0.f;
    float* p = stack + b * (long)LL * C_STACK + c;
    for (int l = 0; l < LL; ++l) {
        acc += *p;
        *p = acc;
        p += C_STACK;
    }
}

// ---------------------------------------------------------------------------
// state/match/out_c build -> bf16 (GEMM A operand).  Thread per (b,l,k,h).
// ---------------------------------------------------------------------------
__global__ void outc_kernel(const float* __restrict__ stack,
                            const float* __restrict__ q,
                            const float* __restrict__ norm_scale,
                            __bf16* __restrict__ out_c)
{
    long idx = (long)blockIdx.x * blockDim.x + threadIdx.x;
    long total = (long)NTOK * KK * HH;
    if (idx >= total) return;
    int  h     = (int)(idx % HH);
    int  k     = (int)((idx / HH) % KK);
    long token = idx / (KK * HH);

    const float* cs = stack + token * C_STACK;
    float inv = 1.f / fmaxf(cs[k], 1e-4f);
    float sre = cs[KK + k * HH + h] * inv;
    float sim = cs[KK + DIM_MEM + k * HH + h] * inv;

    int kq = k / NREP;
    const float* qrow = q + token * DIM_QTOT + kq * (2 * HH) + 2 * h;
    float qre = qrow[0], qim = qrow[1];
    float mre = sre * qre + sim * qim;
    float mim = sim * qre - sre * qim;

    __bf16* oc = out_c + (token * KK + k) * (2 * HH);
    oc[h]      = (__bf16)(mre * norm_scale[k * (2 * HH) + h]);
    oc[HH + h] = (__bf16)(mim * norm_scale[k * (2 * HH) + HH + h]);
}

// ---------------------------------------------------------------------------
// y_spec *= sigmoid(gate_logits[b,l,k]).
// ---------------------------------------------------------------------------
__global__ void gate_mul_kernel(const float* __restrict__ z_aux,
                                float* __restrict__ y_spec)
{
    long idx = (long)blockIdx.x * blockDim.x + threadIdx.x;
    long total = (long)NTOK * KK * DIM_SWH;
    if (idx >= total) return;
    int  k     = (int)((idx / DIM_SWH) % KK);
    long token = idx / ((long)KK * DIM_SWH);
    float g = z_aux[token * DIM_AUX + DIM_QTOT + DIM_LAT + k];
    y_spec[idx] *= sigmoid_f(g);
}

// ---------------------------------------------------------------------------
// Highway combine + SiLU -> bf16 (GEMM A operand).
// ---------------------------------------------------------------------------
__global__ void combine_kernel(const float* __restrict__ y_spec,
                               const float* __restrict__ y_skip,
                               const float* __restrict__ highway_scale,
                               __bf16* __restrict__ y_pre)
{
    constexpr int JH = DIM_SWH / 2;   // 192
    long idx = (long)blockIdx.x * blockDim.x + threadIdx.x;
    long total = (long)NTOK * KK * JH;
    if (idx >= total) return;
    int  j     = (int)(idx % JH);
    int  k     = (int)((idx / JH) % KK);
    long token = idx / ((long)KK * JH);

    float hs = highway_scale[k];
    long base = (token * KK + k) * (long)DIM_SWH;
    float sv = y_spec[base + j],      sg = y_spec[base + JH + j];
    float kv = y_skip[base + j],      kg = y_skip[base + JH + j];
    float val  = sv + kv * hs;
    float gate = sg + kg * hs;
    y_pre[token * DIM_YPRE + k * JH + j] = (__bf16)(val * gate * sigmoid_f(gate));
}

// ---------------------------------------------------------------------------
// Host-side launcher
// ---------------------------------------------------------------------------
extern "C" void kernel_launch(void* const* d_in, const int* in_sizes, int n_in,
                              void* d_out, int out_size, void* d_ws, size_t ws_size,
                              hipStream_t stream)
{
    const float* x            = (const float*)d_in[0];
    const float* W_mem        = (const float*)d_in[1];
    const float* conv_kernel  = (const float*)d_in[2];
    const float* W_aux        = (const float*)d_in[3];
    const float* rms_scale    = (const float*)d_in[4];
    const float* theta_raw    = (const float*)d_in[5];
    const float* tanh_scale   = (const float*)d_in[6];
    const float* decay_slopes = (const float*)d_in[7];
    const float* score_scale  = (const float*)d_in[8];
    const float* W_readout    = (const float*)d_in[9];
    const float* norm_scale   = (const float*)d_in[10];
    const float* highway_scale= (const float*)d_in[11];
    const float* W_skip       = (const float*)d_in[12];
    const float* W_out        = (const float*)d_in[13];
    float*       out          = (float*)d_out;

    const long T = NTOK;

    // ---- fp32 scratch ------------------------------------------------------
    float* ws    = (float*)d_ws;
    float* zmem  = ws;                          // T*780
    float* zconv = zmem  + T * DIM_MEMK;        // T*780
    float* zaux  = zconv + T * DIM_MEMK;        // T*972
    float* qbuf  = zaux  + T * DIM_AUX;         // T*768
    float* stack = qbuf  + T * DIM_QTOT;        // T*1548
    float* yspec = stack + T * C_STACK;         // T*4608
    float* yskip = yspec + T * KK * DIM_SWH;    // T*4608
    float* fend  = yskip + T * KK * DIM_SWH;

    // ---- bf16 scratch ------------------------------------------------------
    __bf16* xbf     = (__bf16*)fend;            // T*768
    __bf16* outcbf  = xbf     + T * DD;         // T*1536
    __bf16* yprebf  = outcbf  + T * 2 * HH * KK;// T*2304
    __bf16* cskipbf = yprebf  + T * DIM_YPRE;   // T*192
    __bf16* wmemT   = cskipbf + T * DIM_LAT;    // 896*768
    __bf16* wauxT   = wmemT  + (long)NPAD_MEM * DD;        // 1024*768
    __bf16* wroT    = wauxT  + (long)NPAD_AUX * DD;        // 12*384*128
    __bf16* wskipT  = wroT   + (long)KK * DIM_SWH * 2*HH;  // 4608*192
    __bf16* woutT   = wskipT + (long)KK * DIM_SWH * DIM_LAT; // 768*2304

    const dim3 blk(256);
    const int GM = NTOK / 64;                   // 256 row tiles

    // ---- operand preparation (bf16 / transpose+pad) ------------------------
    {
        long n = (long)T * DD;
        cvt_bf16_kernel<<<dim3((unsigned)((n + 255) / 256)), blk, 0, stream>>>(x, xbf, n);
    }
    {
        long n = (long)NPAD_MEM * DD;
        wt_transpose_kernel<<<dim3((unsigned)((n + 255) / 256), 1, 1), blk, 0, stream>>>(
            W_mem, wmemT, DD, DIM_MEMK, NPAD_MEM);
    }
    {
        long n = (long)NPAD_AUX * DD;
        wt_transpose_kernel<<<dim3((unsigned)((n + 255) / 256), 1, 1), blk, 0, stream>>>(
            W_aux, wauxT, DD, DIM_AUX, NPAD_AUX);
    }
    {
        long n = (long)DIM_SWH * 2 * HH;        // per k
        wt_transpose_kernel<<<dim3((unsigned)((n + 255) / 256), 1, KK), blk, 0, stream>>>(
            W_readout, wroT, 2 * HH, DIM_SWH, DIM_SWH);
    }
    {
        long n = (long)KK * DIM_SWH * DIM_LAT;
        wt_transpose_kernel<<<dim3((unsigned)((n + 255) / 256), 1, 1), blk, 0, stream>>>(
            W_skip, wskipT, DIM_LAT, KK * DIM_SWH, KK * DIM_SWH);
    }
    {
        long n = (long)DD * DIM_YPRE;
        wt_transpose_kernel<<<dim3((unsigned)((n + 255) / 256), 1, 1), blk, 0, stream>>>(
            W_out, woutT, DIM_YPRE, DD, DD);
    }

    // 1) z_mem = x @ W_mem
    gemm_bf16_wmma<<<dim3(NPAD_MEM / 128, GM, 1), blk, 0, stream>>>(
        xbf, DD, 0, wmemT, DD, 0, zmem, DIM_MEMK, 0, DIM_MEMK, DD);

    // 2) z_aux = x @ W_aux
    gemm_bf16_wmma<<<dim3(NPAD_AUX / 128, GM, 1), blk, 0, stream>>>(
        xbf, DD, 0, wauxT, DD, 0, zaux, DIM_AUX, 0, DIM_AUX, DD);

    // 3) depthwise causal conv
    {
        long n = (long)T * DIM_MEMK;
        conv_dw_kernel<<<dim3((unsigned)((n + 255) / 256)), blk, 0, stream>>>(
            zmem, conv_kernel, zconv);
    }

    // 4) RMSNorm q ; c_skip slice -> bf16
    rms_q_kernel<<<dim3(NTOK), blk, 0, stream>>>(zaux, rms_scale, qbuf);
    {
        long n = (long)T * DIM_LAT;
        cvt_cskip_kernel<<<dim3((unsigned)((n + 255) / 256)), blk, 0, stream>>>(
            zaux, cskipbf);
    }

    // 5) stack build
    {
        long n = (long)T * KK * HH;
        stack_build_kernel<<<dim3((unsigned)((n + 255) / 256)), blk, 0, stream>>>(
            zconv, theta_raw, tanh_scale, decay_slopes, score_scale, stack);
    }

    // 6) cumsum over L
    {
        long n = (long)BB * C_STACK;
        scan_kernel<<<dim3((unsigned)((n + 255) / 256)), blk, 0, stream>>>(stack);
    }

    // 7) out_c (bf16)
    {
        long n = (long)T * KK * HH;
        outc_kernel<<<dim3((unsigned)((n + 255) / 256)), blk, 0, stream>>>(
            stack, qbuf, norm_scale, outcbf);
    }

    // 8) y_spec: batched over k.  A_k = outcbf + k*128 (lda 1536),
    //    B_k = wroT + k*384*128, C_k = yspec + k*384 (ldc 4608)
    gemm_bf16_wmma<<<dim3(DIM_SWH / 128, GM, KK), blk, 0, stream>>>(
        outcbf, KK * 2 * HH, 2 * HH,
        wroT, 2 * HH, (long)DIM_SWH * 2 * HH,
        yspec, KK * DIM_SWH, DIM_SWH,
        DIM_SWH, 2 * HH);

    // 9) gate multiply
    {
        long n = (long)T * KK * DIM_SWH;
        gate_mul_kernel<<<dim3((unsigned)((n + 255) / 256)), blk, 0, stream>>>(
            zaux, yspec);
    }

    // 10) y_skip = c_skip @ W_skip
    gemm_bf16_wmma<<<dim3(KK * DIM_SWH / 128, GM, 1), blk, 0, stream>>>(
        cskipbf, DIM_LAT, 0, wskipT, DIM_LAT, 0,
        yskip, KK * DIM_SWH, 0, KK * DIM_SWH, DIM_LAT);

    // 11) highway + SiLU combine (bf16)
    {
        long n = (long)T * KK * (DIM_SWH / 2);
        combine_kernel<<<dim3((unsigned)((n + 255) / 256)), blk, 0, stream>>>(
            yspec, yskip, highway_scale, yprebf);
    }

    // 12) final: out = y_pre @ W_out
    gemm_bf16_wmma<<<dim3(DD / 128, GM, 1), blk, 0, stream>>>(
        yprebf, DIM_YPRE, 0, woutT, DIM_YPRE, 0,
        out, DD, 0, DD, DIM_YPRE);
}
